// SecureAttention_86603720556606
// MI455X (gfx1250) — compile-verified, matrix operand order checked by reading, not run on hardware
//
#include <hip/hip_runtime.h>
#include <hip/hip_bf16.h>

// ---------------------------------------------------------------------------
// CDNA5 (gfx1250) wave32 WMMA attention pipeline, bf16 operands / f32 accum.
// Async-tensor staging (global_load_async_to_lds_b128) for attention tiles.
// ---------------------------------------------------------------------------

typedef __attribute__((ext_vector_type(16))) __bf16 bf16x16;
typedef __attribute__((ext_vector_type(8)))  float  f32x8;

union FragB {
  bf16x16 v;
  uint4   q[2];
};

__device__ __forceinline__ f32x8 wmma_bf16(bf16x16 a, bf16x16 b, f32x8 c) {
  // (neg_a, A, neg_b, B, c_mod, C, reuse_a, reuse_b)
  return __builtin_amdgcn_wmma_f32_16x16x32_bf16(false, a, false, b, (short)0, c,
                                                 false, false);
}

// Async copy 16B global -> LDS per lane; tracked on ASYNCcnt.
__device__ __forceinline__ void async_b128_to_lds(void* lds_ptr, const void* gptr) {
  const unsigned lds_addr = (unsigned)(size_t)lds_ptr;  // flat[31:0] == LDS offset
  asm volatile("global_load_async_to_lds_b128 %0, %1, off"
               :: "v"(lds_addr), "v"(gptr) : "memory");
}
__device__ __forceinline__ void wait_async0() {
  asm volatile("s_wait_asynccnt 0x0" ::: "memory");
}

#define ATT_B  2
#define ATT_S  2048
#define ATT_D  1024
#define ATT_H  16
#define ATT_HD 64

// ---------------------------------------------------------------------------
// GEMM: C[M,N] = A[M,K] @ W[N,K]^T + bias
//   IN_F32  : A is fp32 (converted to bf16 in LDS), else A is bf16
//   STORE_T : store result transposed per-batch: Out[b][n][s] (bf16)  (for V)
//   OUT_F32 : store fp32 row-major (final projection into d_out)
// 128x128 block tile, 8 waves (2M x 4N), each wave 64x32 = 4x2 WMMA tiles.
// ---------------------------------------------------------------------------
#define GT_M 128
#define GT_N 128
#define GT_K 32
#define LDSTR 40  // bf16 per LDS row: 32 data + 8 pad (80B, bank-conflict-free)

template<int IN_F32, int STORE_T, int OUT_F32>
__global__ __launch_bounds__(256) void gemm_wt(
    const void* __restrict__ Ain, const float* __restrict__ W,
    const float* __restrict__ bias, void* __restrict__ Out,
    int M, int N, int Kd, int Sdim, int Dtot)
{
  __shared__ __align__(16) __bf16 lA[GT_M * LDSTR];
  __shared__ __align__(16) __bf16 lB[GT_N * LDSTR];

  const int tid      = threadIdx.x;
  const int wave     = tid >> 5;
  const int lane     = tid & 31;
  const int lane16   = lane & 15;
  const int laneHalf = lane >> 4;
  const int m0 = blockIdx.y * GT_M;
  const int n0 = blockIdx.x * GT_N;
  const int wm = (wave >> 2) * 64;   // wave M offset within tile
  const int wn = (wave & 3) * 32;    // wave N offset within tile

  f32x8 acc[4][2];
#pragma unroll
  for (int i = 0; i < 4; i++)
#pragma unroll
    for (int j = 0; j < 2; j++)
#pragma unroll
      for (int e = 0; e < 8; e++) acc[i][j][e] = 0.f;

  const int  ldr = tid & 127;    // row each thread stages
  const bool isA = tid < 128;    // threads 0-127: A rows, 128-255: W rows

  for (int k0 = 0; k0 < Kd; k0 += GT_K) {
    // ---- cooperative stage (fp32 -> bf16 convert) into padded LDS ----
    __bf16* dst = (isA ? lA : lB) + ldr * LDSTR;
    if (isA && !IN_F32) {
      const uint4* s4 = (const uint4*)((const __bf16*)Ain + (size_t)(m0 + ldr) * Kd + k0);
      uint4* d4 = (uint4*)dst;
#pragma unroll
      for (int i = 0; i < 4; i++) d4[i] = s4[i];
    } else {
      const float* srow = isA ? ((const float*)Ain + (size_t)(m0 + ldr) * Kd + k0)
                              : (W + (size_t)(n0 + ldr) * Kd + k0);
      const float4* s4 = (const float4*)srow;
#pragma unroll
      for (int i = 0; i < 8; i++) {
        float4 f = s4[i];
        dst[i * 4 + 0] = (__bf16)f.x;
        dst[i * 4 + 1] = (__bf16)f.y;
        dst[i * 4 + 2] = (__bf16)f.z;
        dst[i * 4 + 3] = (__bf16)f.w;
      }
    }
    if (k0 + GT_K < Kd) {  // prefetch next K-slab (global_prefetch_b8)
      const char* pf = isA
        ? (IN_F32 ? (const char*)((const float*)Ain + (size_t)(m0 + ldr) * Kd + k0 + GT_K)
                  : (const char*)((const __bf16*)Ain + (size_t)(m0 + ldr) * Kd + k0 + GT_K))
        : (const char*)(W + (size_t)(n0 + ldr) * Kd + k0 + GT_K);
      __builtin_prefetch(pf, 0, 1);
    }
    __syncthreads();

    // ---- fragment loads (match documented 16-bit A / B lane layouts) ----
    // lane<16: row lane16, K pairs {0..7} at +0B and {16..23} at +32B
    // lane>=16: same row set, K pairs {8..15} at +16B and {24..31} at +48B
    FragB aF[4], bF[2];
    const int off0 = laneHalf * 16;
#pragma unroll
    for (int i = 0; i < 4; i++) {
      const char* base = (const char*)(lA + (wm + i * 16 + lane16) * LDSTR);
      aF[i].q[0] = *(const uint4*)(base + off0);
      aF[i].q[1] = *(const uint4*)(base + off0 + 32);
    }
#pragma unroll
    for (int j = 0; j < 2; j++) {
      const char* base = (const char*)(lB + (wn + j * 16 + lane16) * LDSTR);
      bF[j].q[0] = *(const uint4*)(base + off0);
      bF[j].q[1] = *(const uint4*)(base + off0 + 32);
    }
#pragma unroll
    for (int i = 0; i < 4; i++)
#pragma unroll
      for (int j = 0; j < 2; j++)
        acc[i][j] = wmma_bf16(aF[i].v, bF[j].v, acc[i][j]);
    __syncthreads();
  }

  // ---- epilogue: bias + store. C layout: VGPR r -> rows {r, r+8}, lanes = N.
#pragma unroll
  for (int j = 0; j < 2; j++) {
    const int col = n0 + wn + j * 16 + lane16;
    const float bv = bias[col];
#pragma unroll
    for (int i = 0; i < 4; i++) {
      const int rowBase = m0 + wm + i * 16 + laneHalf * 8;
      if (OUT_F32) {
        float* o = (float*)Out;
#pragma unroll
        for (int r = 0; r < 8; r++)
          o[(size_t)(rowBase + r) * N + col] = acc[i][j][r] + bv;
      } else if (!STORE_T) {
        __bf16* o = (__bf16*)Out;
#pragma unroll
        for (int r = 0; r < 8; r++)
          o[(size_t)(rowBase + r) * N + col] = (__bf16)(acc[i][j][r] + bv);
      } else {
        // V^T store: Out[b][col][s], 8 consecutive tokens -> one b128 store
        const int b = rowBase / Sdim;
        const int s = rowBase - b * Sdim;
        __bf16 tmp[8];
#pragma unroll
        for (int r = 0; r < 8; r++) tmp[r] = (__bf16)(acc[i][j][r] + bv);
        __bf16* o = (__bf16*)Out + ((size_t)b * Dtot + col) * Sdim + s;
        *(uint4*)o = *(const uint4*)tmp;
      }
    }
  }
}

// ---------------------------------------------------------------------------
// Flash-style secure attention. Per block: 128 queries of one (b,h); 8 waves,
// 16 queries each. Streams keys in steps of 32. Computes S^T = K_tile @ Q^T so
// the score C-fragments repack into the probs A-fragment with pure per-lane
// f32->bf16 packing (no transpose through LDS). K/V tiles staged with
// global_load_async_to_lds_b128 (ASYNCcnt); mask-add term precomputed in LDS.
// ---------------------------------------------------------------------------
#define KSTRIDE 72  // bf16 per lK row: 64 data + 8 pad (144B)
#define VSTRIDE 40  // bf16 per lV row: 32 data + 8 pad (80B)

__global__ __launch_bounds__(256) void attn_kernel(
    const __bf16* __restrict__ Qp, const __bf16* __restrict__ Kp,
    const __bf16* __restrict__ VTp, const float* __restrict__ maskp,
    __bf16* __restrict__ Cp)
{
  __shared__ __align__(16) __bf16 lK[32 * KSTRIDE];  // [key][hd]
  __shared__ __align__(16) __bf16 lV[64 * VSTRIDE];  // [hd][key]  (from V^T)
  __shared__ float lMadd[ATT_S];                     // (1-mask)*-1e4 per key
  __shared__ float wsum[8][32];

  const int tid      = threadIdx.x;
  const int wave     = tid >> 5;
  const int lane     = tid & 31;
  const int lane16   = lane & 15;
  const int laneHalf = lane >> 4;

  const int bh   = blockIdx.y;
  const int bIdx = bh / ATT_H;
  const int hIdx = bh % ATT_H;
  const int q0   = blockIdx.x * 128 + wave * 16;  // wave's query base

  // ---- precompute additive mask term once (broadcast-read in hot loop) ----
#pragma unroll
  for (int i = 0; i < ATT_S / 256; i++) {
    const int key = tid + i * 256;
    lMadd[key] = (1.0f - maskp[bIdx * ATT_S + key]) * -10000.0f;
  }

  // Q fragments as B-operand (lane = query row, VGPRs = hd pairs); resident.
  FragB qf[2];
  {
    const char* qrow = (const char*)(Qp +
        ((size_t)bIdx * ATT_S + q0 + lane16) * ATT_D + hIdx * ATT_HD);
#pragma unroll
    for (int h2 = 0; h2 < 2; h2++) {
      qf[h2].q[0] = *(const uint4*)(qrow + h2 * 64 + laneHalf * 16);
      qf[h2].q[1] = *(const uint4*)(qrow + h2 * 64 + laneHalf * 16 + 32);
    }
  }

  f32x8 cacc[4];
#pragma unroll
  for (int t = 0; t < 4; t++)
#pragma unroll
    for (int e = 0; e < 8; e++) cacc[t][e] = 0.f;
  float rsum = 0.f;

  const int kRow = tid >> 3, kChunk = tid & 7;  // 32 rows x 8 chunks of 16B
  const int vRow = tid >> 2, vChunk = tid & 3;  // 64 rows x 4 chunks of 16B
  char* lKdst = (char*)lK + kRow * (KSTRIDE * 2) + kChunk * 16;
  char* lVdst = (char*)lV + vRow * (VSTRIDE * 2) + vChunk * 16;

  for (int j = 0; j < ATT_S; j += 32) {
    // ---- async stage K tile [32 keys][64 hd] and V^T tile [64 hd][32 keys]
    async_b128_to_lds(lKdst,
        Kp + ((size_t)bIdx * ATT_S + j + kRow) * ATT_D + hIdx * ATT_HD + kChunk * 8);
    async_b128_to_lds(lVdst,
        VTp + ((size_t)bIdx * ATT_D + hIdx * ATT_HD + vRow) * ATT_S + j + vChunk * 8);
    wait_async0();
    __syncthreads();

    // ---- S^T tiles: rows = keys (VGPRs), cols = queries (lanes) ----
    f32x8 st[2];
#pragma unroll
    for (int kk = 0; kk < 2; kk++) {
      FragB kf0, kf1;  // A-operand: lane = key row; hd halves 0..31 / 32..63
      const char* base = (char*)lK + (kk * 16 + lane16) * (KSTRIDE * 2) + laneHalf * 16;
      kf0.q[0] = *(const uint4*)(base);
      kf0.q[1] = *(const uint4*)(base + 32);
      kf1.q[0] = *(const uint4*)(base + 64);
      kf1.q[1] = *(const uint4*)(base + 96);
      f32x8 z;
#pragma unroll
      for (int e = 0; e < 8; e++) z[e] = 0.f;
      z      = wmma_bf16(kf0.v, qf[0].v, z);
      st[kk] = wmma_bf16(kf1.v, qf[1].v, z);
    }

    // ---- scale + mask + secure softmax numerator (elementwise f32) ----
#pragma unroll
    for (int kk = 0; kk < 2; kk++) {
#pragma unroll
      for (int r = 0; r < 8; r++) {
        const int key = j + kk * 16 + laneHalf * 8 + r;
        float s = st[kk][r] * 0.125f + lMadd[key];  // /sqrt(64) + mask term
        float c = fminf(3.0f, fmaxf(-3.0f, s));
        float sig = 0.5f + 0.25f * c - 0.01f * c * c * c;
        sig = fminf(0.99f, fmaxf(0.01f, sig));
        st[kk][r] = sig;
        rsum += sig;
      }
    }

    // ---- repack probs C-frags -> ctx A-frag (pure per-lane packing) ----
    // lane<16 holds keys {0..7} (st[0]) and {16..23} (st[1]) for q=lane;
    // lane>=16 holds keys {8..15} / {24..31} -> exactly the A 16x32 layout.
    FragB pf;
#pragma unroll
    for (int e = 0; e < 8; e++) pf.v[e]     = (__bf16)st[0][e];
#pragma unroll
    for (int e = 0; e < 8; e++) pf.v[8 + e] = (__bf16)st[1][e];

    // ---- ctx += probs @ V : 4 hd tiles; B-operand rows straight from lV ----
#pragma unroll
    for (int t = 0; t < 4; t++) {
      FragB vf;
      const char* base = (char*)lV + (t * 16 + lane16) * (VSTRIDE * 2) + laneHalf * 16;
      vf.q[0] = *(const uint4*)(base);
      vf.q[1] = *(const uint4*)(base + 32);
      cacc[t] = wmma_bf16(pf.v, vf.v, cacc[t]);
    }
    __syncthreads();
  }

  // ---- combine per-lane partial row sums (lane n + lane n+16) via LDS ----
  wsum[wave][lane] = rsum;
  __syncthreads();

#pragma unroll
  for (int r = 0; r < 8; r++) {
    const int qrow = laneHalf * 8 + r;
    const float rs = wsum[wave][qrow] + wsum[wave][qrow + 16] + 1e-8f;
    const float inv = 1.0f / rs;
    const int qg = q0 + qrow;
#pragma unroll
    for (int t = 0; t < 4; t++) {
      const int col = hIdx * ATT_HD + t * 16 + lane16;
      Cp[((size_t)bIdx * ATT_S + qg) * ATT_D + col] = (__bf16)(cacc[t][r] * inv);
    }
  }
}

// ---------------------------------------------------------------------------
extern "C" void kernel_launch(void* const* d_in, const int* in_sizes, int n_in,
                              void* d_out, int out_size, void* d_ws, size_t ws_size,
                              hipStream_t stream) {
  const float* hs = (const float*)d_in[0];
  const float* am = (const float*)d_in[1];
  const float* Wq = (const float*)d_in[2];
  const float* bq = (const float*)d_in[3];
  const float* Wk = (const float*)d_in[4];
  const float* bk = (const float*)d_in[5];
  const float* Wv = (const float*)d_in[6];
  const float* bv = (const float*)d_in[7];
  const float* Wo = (const float*)d_in[8];
  const float* bo = (const float*)d_in[9];

  char* ws = (char*)d_ws;
  const size_t SEG = (size_t)ATT_B * ATT_S * ATT_D * sizeof(__bf16);  // 8 MB
  __bf16* Qb = (__bf16*)(ws + 0 * SEG);   // [B*S, D] bf16
  __bf16* Kb = (__bf16*)(ws + 1 * SEG);   // [B*S, D] bf16
  __bf16* VT = (__bf16*)(ws + 2 * SEG);   // [B, D, S] bf16 (transposed V)
  __bf16* Cb = (__bf16*)(ws + 3 * SEG);   // [B*S, D] bf16 (attention ctx)

  const int M = ATT_B * ATT_S;  // 4096
  const int N = ATT_D, Kd = ATT_D;
  dim3 gG(N / GT_N, M / GT_M);  // (8, 32)

  gemm_wt<1, 0, 0><<<gG, 256, 0, stream>>>(hs, Wq, bq, Qb, M, N, Kd, ATT_S, ATT_D);
  gemm_wt<1, 0, 0><<<gG, 256, 0, stream>>>(hs, Wk, bk, Kb, M, N, Kd, ATT_S, ATT_D);
  gemm_wt<1, 1, 0><<<gG, 256, 0, stream>>>(hs, Wv, bv, VT, M, N, Kd, ATT_S, ATT_D);

  dim3 gA(ATT_S / 128, ATT_B * ATT_H);  // (16, 32)
  attn_kernel<<<gA, 256, 0, stream>>>(Qb, Kb, VT, am, Cb);

  gemm_wt<0, 0, 1><<<gG, 256, 0, stream>>>(Cb, Wo, bo, d_out, M, N, Kd, ATT_S, ATT_D);

  (void)in_sizes; (void)n_in; (void)out_size; (void)ws_size;
}
